// MMTDecoderBlock_10952166605367
// MI455X (gfx1250) — matrix operand order repaired and checked by reading, not compiled, not run on hardware
//
#include <hip/hip_runtime.h>
#include <hip/hip_bf16.h>

// ---------------------------------------------------------------------------
// MMT decoder block for MI455X (gfx1250, wave32, WMMA 16x16x32 bf16)
// ---------------------------------------------------------------------------

typedef unsigned short u16;
typedef __attribute__((ext_vector_type(16))) __bf16  v16bf;
typedef __attribute__((ext_vector_type(8)))  float   v8f;
typedef __attribute__((ext_vector_type(4)))  unsigned int u32x4;

union ABFrag { u32x4 q[2]; v16bf v; };

#define TOK   150528            // B*NC*H*W = total tokens
#define CDIM  128
#define HEADS 4
#define HD    32
#define NTOKW 147               // NC * 49 tokens per window
#define NPAD  160               // padded to 10 wmma tiles
#define SCALE_Q 0.17677669529663689f   // 32^-0.5

__device__ __forceinline__ u16 f2bf(float f) {
  unsigned u = __builtin_bit_cast(unsigned, f);
  unsigned r = u + 0x7fffu + ((u >> 16) & 1u);
  return (u16)(r >> 16);
}

__device__ __forceinline__ float wave_sum(float v) {
  v += __shfl_xor(v, 16);
  v += __shfl_xor(v, 8);
  v += __shfl_xor(v, 4);
  v += __shfl_xor(v, 2);
  v += __shfl_xor(v, 1);
  return v;
}

// region id for shifted-window mask (image coordinate in rolled frame)
__device__ __forceinline__ int sw_region(int h) {
  return (h < 105) ? 0 : ((h < 109) ? 1 : 2);
}

// decode window-layout token m -> source index (natural layout, after roll -3)
__device__ __forceinline__ long win_src_index(int m, int* nch_out) {
  int b_  = m / NTOKW;
  int t   = m - b_ * NTOKW;
  int nch = t / 49;
  int pos = t - nch * 49;
  int r   = pos / 7, cc = pos - r * 7;
  int win = b_ & 255;
  int b   = b_ >> 8;
  int wh  = win >> 4, ww = win & 15;
  int hs  = (wh * 7 + r + 3) % 112;
  int ws  = (ww * 7 + cc + 3) % 112;
  *nch_out = nch;
  return ((((long)(b * 3 + nch)) * 112 + hs) * 112 + ws) * CDIM;
}

// ---------------------------------------------------------------------------
// weight convert: f32 [K x N] row-major  ->  bf16 transposed [N x K] row-major
// ---------------------------------------------------------------------------
__global__ __launch_bounds__(256)
void cvt_t_kernel(const float* __restrict__ s, u16* __restrict__ d, int K, int N) {
  int i = blockIdx.x * 256 + threadIdx.x;
  if (i < K * N) {
    int k = i / N, n = i - k * N;
    d[(long)n * K + k] = f2bf(s[i]);
  }
}

// ---------------------------------------------------------------------------
// prep1: LN1 + roll(-3,-3) + window partition ; xwe = ln + ct, xw = ln
// ---------------------------------------------------------------------------
__global__ __launch_bounds__(256)
void prep1_kernel(const float* __restrict__ x,
                  const float* __restrict__ g, const float* __restrict__ bta,
                  const float* __restrict__ ct,
                  u16* __restrict__ xwe, u16* __restrict__ xw) {
  __shared__ float redA[2][4], redB[2][4];
  int tid = threadIdx.x;
  int sub = tid >> 7, wv = (tid >> 5) & 3, c = tid & 127;
  int m = blockIdx.x * 2 + sub;
  int nch;
  long src = win_src_index(m, &nch);
  float v = x[src + c];
  float s = wave_sum(v);
  if ((tid & 31) == 0) redA[sub][wv] = s;
  __syncthreads();
  float mean = (redA[sub][0] + redA[sub][1] + redA[sub][2] + redA[sub][3]) * (1.0f / 128.0f);
  float d = v - mean;
  float s2 = wave_sum(d * d);
  if ((tid & 31) == 0) redB[sub][wv] = s2;
  __syncthreads();
  float var = (redB[sub][0] + redB[sub][1] + redB[sub][2] + redB[sub][3]) * (1.0f / 128.0f);
  float ln = d * rsqrtf(var + 1e-5f) * g[c] + bta[c];
  long o = (long)m * CDIM + c;
  xw[o]  = f2bf(ln);
  xwe[o] = f2bf(ln + ct[nch * CDIM + c]);
}

// ---------------------------------------------------------------------------
// prep2: qin = LN2(x1)|window + ct ; kin = x_kv|window + ce ; vin = x_kv|window
// ---------------------------------------------------------------------------
__global__ __launch_bounds__(256)
void prep2_kernel(const float* __restrict__ x1,
                  const float* __restrict__ g, const float* __restrict__ bta,
                  const float* __restrict__ ct, const float* __restrict__ ce,
                  const float* __restrict__ xkv,
                  u16* __restrict__ qin, u16* __restrict__ kin, u16* __restrict__ vin) {
  __shared__ float redA[2][4], redB[2][4];
  int tid = threadIdx.x;
  int sub = tid >> 7, wv = (tid >> 5) & 3, c = tid & 127;
  int m = blockIdx.x * 2 + sub;
  int nch;
  long src = win_src_index(m, &nch);
  float v = x1[src + c];
  float s = wave_sum(v);
  if ((tid & 31) == 0) redA[sub][wv] = s;
  __syncthreads();
  float mean = (redA[sub][0] + redA[sub][1] + redA[sub][2] + redA[sub][3]) * (1.0f / 128.0f);
  float d = v - mean;
  float s2 = wave_sum(d * d);
  if ((tid & 31) == 0) redB[sub][wv] = s2;
  __syncthreads();
  float var = (redB[sub][0] + redB[sub][1] + redB[sub][2] + redB[sub][3]) * (1.0f / 128.0f);
  float ln = d * rsqrtf(var + 1e-5f) * g[c] + bta[c];
  float kv = xkv[src + c];
  long o = (long)m * CDIM + c;
  qin[o] = f2bf(ln + ct[nch * CDIM + c]);
  kin[o] = f2bf(kv + ce[nch * CDIM + c]);
  vin[o] = f2bf(kv);
}

// ---------------------------------------------------------------------------
// prep3: LN3(x2) natural layout -> bf16
// ---------------------------------------------------------------------------
__global__ __launch_bounds__(256)
void prep3_kernel(const float* __restrict__ x2,
                  const float* __restrict__ g, const float* __restrict__ bta,
                  u16* __restrict__ out) {
  __shared__ float redA[2][4], redB[2][4];
  int tid = threadIdx.x;
  int sub = tid >> 7, wv = (tid >> 5) & 3, c = tid & 127;
  int m = blockIdx.x * 2 + sub;
  long src = (long)m * CDIM;
  float v = x2[src + c];
  float s = wave_sum(v);
  if ((tid & 31) == 0) redA[sub][wv] = s;
  __syncthreads();
  float mean = (redA[sub][0] + redA[sub][1] + redA[sub][2] + redA[sub][3]) * (1.0f / 128.0f);
  float d = v - mean;
  float s2 = wave_sum(d * d);
  if ((tid & 31) == 0) redB[sub][wv] = s2;
  __syncthreads();
  float var = (redB[sub][0] + redB[sub][1] + redB[sub][2] + redB[sub][3]) * (1.0f / 128.0f);
  out[src + c] = f2bf(d * rsqrtf(var + 1e-5f) * g[c] + bta[c]);
}

// ---------------------------------------------------------------------------
// Generic bf16 WMMA GEMM:  out = epilogue( A[MxK] @ WT^T + bias )
// A row-major [M x K], WT row-major [N x K] (pre-transposed weights).
// Block tile 128x128, 8 waves (each 32x64 = 2x4 wmma tiles), K step 32.
// Double-buffered LDS staging: one barrier per K-step; next tile's global
// loads are issued before computing the current tile, so HBM/L2 latency
// overlaps the WMMA chain.
// MODE: 0 = (acc+bias)*scale -> bf16
//       1 = gelu(acc+bias)   -> bf16
//       2 = window-reverse + roll(+3) scatter: f32 out = res + acc + bias
//       3 = f32 out = res + acc + bias (plain layout)
// ---------------------------------------------------------------------------
template <int MODE>
__global__ __launch_bounds__(256)
void gemm_bf16_kernel(const u16* __restrict__ A, const u16* __restrict__ WT,
                      const float* __restrict__ bias, float scale,
                      int M, int N, int K,
                      u16* __restrict__ outb, const float* __restrict__ res,
                      float* __restrict__ outf) {
  __shared__ u16 As[2][128 * 40];   // A tiles, row-major, stride 40 (pad)
  __shared__ u16 Bt[2][128 * 40];   // B tiles [n][k], stride 40

  int tid  = threadIdx.x;
  long row0 = (long)blockIdx.y * 128;
  int col0 = blockIdx.x * 128;

  int wave = tid >> 5, lane = tid & 31;
  int l15 = lane & 15, hi = lane >> 4;
  int wm = wave >> 1, wn = wave & 1;
  int kq = hi ? 8 : 0;     // A-frag K base (CDNA5 16-bit A layout)
  int kb = hi ? 8 : 0;     // B-frag K base

  auto stage = [&](int kt, u16* AsB, u16* BtB) {
#pragma unroll
    for (int s = 0; s < 2; ++s) {
      int id = tid + s * 256;
      int r = id >> 2, seg = id & 3;
      *(u32x4*)(AsB + r * 40 + seg * 8) =
          *(const u32x4*)(A + (row0 + r) * K + kt + seg * 8);
      *(u32x4*)(BtB + r * 40 + seg * 8) =
          *(const u32x4*)(WT + (long)(col0 + r) * K + kt + seg * 8);
    }
  };

  v8f acc[2][4];
#pragma unroll
  for (int i = 0; i < 2; ++i)
#pragma unroll
    for (int j = 0; j < 4; ++j)
      acc[i][j] = (v8f){0.f, 0.f, 0.f, 0.f, 0.f, 0.f, 0.f, 0.f};

  int nk = K >> 5;
  stage(0, As[0], Bt[0]);

  for (int it = 0; it < nk; ++it) {
    __syncthreads();                       // tile 'it' staged; buffer it^1 free
    if (it + 1 < nk)
      stage((it + 1) << 5, As[(it + 1) & 1], Bt[(it + 1) & 1]);

    const u16* Ac = As[it & 1];
    const u16* Bc = Bt[it & 1];
    ABFrag a[2];
#pragma unroll
    for (int i = 0; i < 2; ++i) {
      int r = wm * 32 + i * 16 + l15;
      a[i].q[0] = *(const u32x4*)(Ac + r * 40 + kq);
      a[i].q[1] = *(const u32x4*)(Ac + r * 40 + kq + 16);
    }
#pragma unroll
    for (int j = 0; j < 4; ++j) {
      ABFrag b;
      int cn = wn * 64 + j * 16 + l15;
      b.q[0] = *(const u32x4*)(Bc + cn * 40 + kb);
      b.q[1] = *(const u32x4*)(Bc + cn * 40 + kb + 16);
      acc[0][j] = __builtin_amdgcn_wmma_f32_16x16x32_bf16(
          false, a[0].v, false, b.v, (short)0, acc[0][j], false, false);
      acc[1][j] = __builtin_amdgcn_wmma_f32_16x16x32_bf16(
          false, a[1].v, false, b.v, (short)0, acc[1][j], false, false);
    }
  }

  // epilogue (fully unrolled, row math hoisted out of the column loop)
#pragma unroll
  for (int i = 0; i < 2; ++i) {
#pragma unroll
    for (int e = 0; e < 8; ++e) {
      long grow = row0 + wm * 32 + i * 16 + e + hi * 8;
      long dst = 0;
      if (MODE == 2) {
        int nch;
        dst = win_src_index((int)grow, &nch);   // same map as the gather side
      }
#pragma unroll
      for (int j = 0; j < 4; ++j) {
        int gcol = col0 + wn * 64 + j * 16 + l15;
        float v = (acc[i][j][e] + bias[gcol]) * scale;
        if (MODE == 0) {
          outb[grow * N + gcol] = f2bf(v);
        } else if (MODE == 1) {
          v = 0.5f * v * (1.0f + erff(v * 0.70710678118654752f));
          outb[grow * N + gcol] = f2bf(v);
        } else if (MODE == 2) {
          outf[dst + gcol] = res[dst + gcol] + v;
        } else {
          outf[grow * N + gcol] = res[grow * N + gcol] + v;
        }
      }
    }
  }
}

// ---------------------------------------------------------------------------
// Fused window attention, one block per (window, head). 10 waves (320 thr).
// scores = Q@K^T (+rel-pos bias + shift mask), softmax, out = P@V.
// ---------------------------------------------------------------------------
__global__ __launch_bounds__(320)
void attn_kernel(const u16* __restrict__ Q, const u16* __restrict__ K,
                 const u16* __restrict__ V, const float* __restrict__ rpb,
                 u16* __restrict__ O) {
  extern __shared__ u16 smem[];
  u16* Qs = smem;                       // 160*32
  u16* Ks = Qs + NPAD * 32;             // 160*32
  u16* Vt = Ks + NPAD * 32;             // 32*168
  u16* Ps = Vt + 32 * 168;              // 160*168

  int tid = threadIdx.x;
  int b_  = blockIdx.x >> 2;
  int h   = blockIdx.x & 3;
  long base = ((long)b_ * NTOKW) * CDIM + h * HD;

  // stage Q, K (exactly one (row,seg) pair per thread: NPAD*2 == 320)
  {
    int row = tid >> 1, seg = tid & 1;
    if (row < NTOKW) {
      *(u32x4*)(Qs + row * 32 + seg * 16) =
          *(const u32x4*)(Q + base + (long)row * CDIM + seg * 16);
      *(u32x4*)(Ks + row * 32 + seg * 16) =
          *(const u32x4*)(K + base + (long)row * CDIM + seg * 16);
    } else {
      u32x4 z = (u32x4){0u, 0u, 0u, 0u};
      *(u32x4*)(Qs + row * 32 + seg * 16) = z;
      *(u32x4*)(Ks + row * 32 + seg * 16) = z;
    }
  }
  // stage V transposed [hd][key]
#pragma unroll
  for (int s = 0; s < 8; ++s) {
    int i = tid + s * 320;
    int row = i >> 4;
    int dp  = (i & 15) << 1;
    unsigned uv = (row < NTOKW)
        ? *(const unsigned*)(V + base + (long)row * CDIM + dp) : 0u;
    Vt[dp * 168 + row]       = (u16)(uv & 0xffffu);
    Vt[(dp + 1) * 168 + row] = (u16)(uv >> 16);
  }
  __syncthreads();

  int wave = tid >> 5, lane = tid & 31;
  int l15 = lane & 15, hi = lane >> 4;
  int kq = hi ? 8 : 0;
  int kb = hi ? 8 : 0;
  int rowb = wave * 16;

  int win = b_ & 255;
  int wh = win >> 4, ww = win & 15;

  // scores: one wmma per 16x16 tile (K=32 fits a single instruction)
  ABFrag a;
  a.q[0] = *(const u32x4*)(Qs + (rowb + l15) * 32 + kq);
  a.q[1] = *(const u32x4*)(Qs + (rowb + l15) * 32 + kq + 16);
  v8f acc[10];
  v8f zero = (v8f){0.f, 0.f, 0.f, 0.f, 0.f, 0.f, 0.f, 0.f};
#pragma unroll
  for (int j = 0; j < 10; ++j) {
    ABFrag b;
    b.q[0] = *(const u32x4*)(Ks + (j * 16 + l15) * 32 + kb);
    b.q[1] = *(const u32x4*)(Ks + (j * 16 + l15) * 32 + kb + 16);
    acc[j] = __builtin_amdgcn_wmma_f32_16x16x32_bf16(
        false, a.v, false, b.v, (short)0, zero, false, false);
  }

  // rel-pos bias + shift mask (analytic, no tables); row math hoisted
#pragma unroll
  for (int e = 0; e < 8; ++e) {
    int grow = rowb + e + hi * 8;
    int pi = grow % 49;
    int ri = pi / 7, ci = pi - ri * 7;
    int regi = sw_region(wh * 7 + ri) * 3 + sw_region(ww * 7 + ci);
    bool rowok = (grow < NTOKW);
#pragma unroll
    for (int j = 0; j < 10; ++j) {
      int col = j * 16 + l15;
      float sc = acc[j][e];
      if (rowok && col < NTOKW) {
        int pj = col % 49;
        int rj = pj / 7, cj = pj - rj * 7;
        int rpidx = ((ri - rj) + 6) * 13 + ((ci - cj) + 6);
        sc += rpb[rpidx * HEADS + h];
        int regj = sw_region(wh * 7 + rj) * 3 + sw_region(ww * 7 + cj);
        if (regi != regj) sc -= 100.0f;
      } else {
        sc = -1e30f;
      }
      acc[j][e] = sc;
    }
  }

  // softmax per row (row lives in one 16-lane half -> xor masks 1,2,4,8)
#pragma unroll
  for (int e = 0; e < 8; ++e) {
    float mx = -1e30f;
#pragma unroll
    for (int j = 0; j < 10; ++j) mx = fmaxf(mx, acc[j][e]);
    mx = fmaxf(mx, __shfl_xor(mx, 1));
    mx = fmaxf(mx, __shfl_xor(mx, 2));
    mx = fmaxf(mx, __shfl_xor(mx, 4));
    mx = fmaxf(mx, __shfl_xor(mx, 8));
    float sum = 0.f;
#pragma unroll
    for (int j = 0; j < 10; ++j) {
      float ex = __expf(acc[j][e] - mx);
      acc[j][e] = ex;
      sum += ex;
    }
    sum += __shfl_xor(sum, 1);
    sum += __shfl_xor(sum, 2);
    sum += __shfl_xor(sum, 4);
    sum += __shfl_xor(sum, 8);
    float inv = 1.0f / sum;
    int grow = rowb + e + hi * 8;
    bool rowok = (grow < NTOKW);
#pragma unroll
    for (int j = 0; j < 10; ++j) {
      int col = j * 16 + l15;
      float p = acc[j][e] * inv;
      if (!rowok || col >= NTOKW) p = 0.f;
      Ps[grow * 168 + col] = f2bf(p);
    }
  }
  __syncthreads();

  // out = P @ V   (rows 16 per wave, 32 cols = HD, K over 160 keys)
  v8f o0 = zero, o1 = zero;
#pragma unroll
  for (int kt = 0; kt < 5; ++kt) {
    ABFrag pa;
    pa.q[0] = *(const u32x4*)(Ps + (rowb + l15) * 168 + kt * 32 + kq);
    pa.q[1] = *(const u32x4*)(Ps + (rowb + l15) * 168 + kt * 32 + kq + 16);
    ABFrag vb0, vb1;
    vb0.q[0] = *(const u32x4*)(Vt + l15 * 168 + kt * 32 + kb);
    vb0.q[1] = *(const u32x4*)(Vt + l15 * 168 + kt * 32 + kb + 16);
    vb1.q[0] = *(const u32x4*)(Vt + (16 + l15) * 168 + kt * 32 + kb);
    vb1.q[1] = *(const u32x4*)(Vt + (16 + l15) * 168 + kt * 32 + kb + 16);
    o0 = __builtin_amdgcn_wmma_f32_16x16x32_bf16(
        false, pa.v, false, vb0.v, (short)0, o0, false, false);
    o1 = __builtin_amdgcn_wmma_f32_16x16x32_bf16(
        false, pa.v, false, vb1.v, (short)0, o1, false, false);
  }
#pragma unroll
  for (int e = 0; e < 8; ++e) {
    int grow = rowb + e + hi * 8;
    if (grow < NTOKW) {
      long ob = ((long)b_ * NTOKW + grow) * CDIM + h * HD;
      O[ob + l15]      = f2bf(o0[e]);
      O[ob + 16 + l15] = f2bf(o1[e]);
    }
  }
}

// ---------------------------------------------------------------------------
// host-side launcher
// ---------------------------------------------------------------------------
extern "C" void kernel_launch(void* const* d_in, const int* in_sizes, int n_in,
                              void* d_out, int out_size, void* d_ws, size_t ws_size,
                              hipStream_t stream) {
  const float* x    = (const float*)d_in[0];
  const float* x_kv = (const float*)d_in[1];
  const float* ct   = (const float*)d_in[2];
  const float* ce   = (const float*)d_in[3];
  const float* n1g = (const float*)d_in[4];
  const float* n1b = (const float*)d_in[5];
  const float* n2g = (const float*)d_in[6];
  const float* n2b = (const float*)d_in[7];
  const float* n3g = (const float*)d_in[8];
  const float* n3b = (const float*)d_in[9];
  const float* sa_wq = (const float*)d_in[10];
  const float* sa_bq = (const float*)d_in[11];
  const float* sa_wk = (const float*)d_in[12];
  const float* sa_bk = (const float*)d_in[13];
  const float* sa_wv = (const float*)d_in[14];
  const float* sa_bv = (const float*)d_in[15];
  const float* sa_wp = (const float*)d_in[16];
  const float* sa_bp = (const float*)d_in[17];
  const float* sa_rpb = (const float*)d_in[18];
  const float* ca_wq = (const float*)d_in[19];
  const float* ca_bq = (const float*)d_in[20];
  const float* ca_wk = (const float*)d_in[21];
  const float* ca_bk = (const float*)d_in[22];
  const float* ca_wv = (const float*)d_in[23];
  const float* ca_bv = (const float*)d_in[24];
  const float* ca_wp = (const float*)d_in[25];
  const float* ca_bp = (const float*)d_in[26];
  const float* ca_rpb = (const float*)d_in[27];
  const float* m_w1 = (const float*)d_in[28];
  const float* m_b1 = (const float*)d_in[29];
  const float* m_w2 = (const float*)d_in[30];
  const float* m_b2 = (const float*)d_in[31];

  // workspace layout
  const size_t BUF = (size_t)TOK * CDIM * sizeof(u16);   // 38.5 MB bf16 buffer
  char* ws = (char*)d_ws;
  u16*  b0 = (u16*)(ws + 0 * BUF);
  u16*  b1 = (u16*)(ws + 1 * BUF);
  u16*  b2 = (u16*)(ws + 2 * BUF);
  u16*  b3 = (u16*)(ws + 3 * BUF);
  u16*  b4 = (u16*)(ws + 4 * BUF);
  u16*  b5 = (u16*)(ws + 5 * BUF);
  float* x1 = (float*)(ws + 6 * BUF);
  float* x2 = (float*)((char*)x1 + (size_t)TOK * CDIM * sizeof(float));
  u16*  wbf = (u16*)((char*)x2 + (size_t)TOK * CDIM * sizeof(float));

  u16* w_sa_q = wbf + 0 * 16384;
  u16* w_sa_k = wbf + 1 * 16384;
  u16* w_sa_v = wbf + 2 * 16384;
  u16* w_sa_p = wbf + 3 * 16384;
  u16* w_ca_q = wbf + 4 * 16384;
  u16* w_ca_k = wbf + 5 * 16384;
  u16* w_ca_v = wbf + 6 * 16384;
  u16* w_ca_p = wbf + 7 * 16384;
  u16* w_m1   = wbf + 8 * 16384;            // transposed 512x128
  u16* w_m2   = w_m1 + 128 * 512;           // transposed 128x512

  // convert all weights to bf16, pre-transposed to [N x K]
  cvt_t_kernel<<<64, 256, 0, stream>>>(sa_wq, w_sa_q, 128, 128);
  cvt_t_kernel<<<64, 256, 0, stream>>>(sa_wk, w_sa_k, 128, 128);
  cvt_t_kernel<<<64, 256, 0, stream>>>(sa_wv, w_sa_v, 128, 128);
  cvt_t_kernel<<<64, 256, 0, stream>>>(sa_wp, w_sa_p, 128, 128);
  cvt_t_kernel<<<64, 256, 0, stream>>>(ca_wq, w_ca_q, 128, 128);
  cvt_t_kernel<<<64, 256, 0, stream>>>(ca_wk, w_ca_k, 128, 128);
  cvt_t_kernel<<<64, 256, 0, stream>>>(ca_wv, w_ca_v, 128, 128);
  cvt_t_kernel<<<64, 256, 0, stream>>>(ca_wp, w_ca_p, 128, 128);
  cvt_t_kernel<<<256, 256, 0, stream>>>(m_w1, w_m1, 128, 512);
  cvt_t_kernel<<<256, 256, 0, stream>>>(m_w2, w_m2, 512, 128);

  const dim3 gGemm128(1, TOK / 128);     // N=128
  const dim3 gGemm512(4, TOK / 128);     // N=512
  const int  PREP_GRID = TOK / 2;
  const size_t ATTN_LDS = (size_t)(NPAD * 32 + NPAD * 32 + 32 * 168 + NPAD * 168) * sizeof(u16);

  // ---- stage 1: shifted-window self attention --------------------------
  prep1_kernel<<<PREP_GRID, 256, 0, stream>>>(x, n1g, n1b, ct, b0, b1);
  gemm_bf16_kernel<0><<<gGemm128, 256, 0, stream>>>(b0, w_sa_q, sa_bq, SCALE_Q,
      TOK, 128, 128, b2, nullptr, nullptr);
  gemm_bf16_kernel<0><<<gGemm128, 256, 0, stream>>>(b0, w_sa_k, sa_bk, 1.0f,
      TOK, 128, 128, b3, nullptr, nullptr);
  gemm_bf16_kernel<0><<<gGemm128, 256, 0, stream>>>(b1, w_sa_v, sa_bv, 1.0f,
      TOK, 128, 128, b4, nullptr, nullptr);
  attn_kernel<<<1024 * HEADS, 320, ATTN_LDS, stream>>>(b2, b3, b4, sa_rpb, b5);
  gemm_bf16_kernel<2><<<gGemm128, 256, 0, stream>>>(b5, w_sa_p, sa_bp, 1.0f,
      TOK, 128, 128, nullptr, x, x1);

  // ---- stage 2: shifted-window cross attention -------------------------
  prep2_kernel<<<PREP_GRID, 256, 0, stream>>>(x1, n2g, n2b, ct, ce, x_kv, b0, b1, b2);
  gemm_bf16_kernel<0><<<gGemm128, 256, 0, stream>>>(b0, w_ca_q, ca_bq, SCALE_Q,
      TOK, 128, 128, b3, nullptr, nullptr);
  gemm_bf16_kernel<0><<<gGemm128, 256, 0, stream>>>(b1, w_ca_k, ca_bk, 1.0f,
      TOK, 128, 128, b4, nullptr, nullptr);
  gemm_bf16_kernel<0><<<gGemm128, 256, 0, stream>>>(b2, w_ca_v, ca_bv, 1.0f,
      TOK, 128, 128, b5, nullptr, nullptr);
  attn_kernel<<<1024 * HEADS, 320, ATTN_LDS, stream>>>(b3, b4, b5, ca_rpb, b0);
  gemm_bf16_kernel<2><<<gGemm128, 256, 0, stream>>>(b0, w_ca_p, ca_bp, 1.0f,
      TOK, 128, 128, nullptr, x1, x2);

  // ---- stage 3: MLP ----------------------------------------------------
  prep3_kernel<<<PREP_GRID, 256, 0, stream>>>(x2, n3g, n3b, b4);
  // hidden (TOK x 512 bf16) spans b0..b3
  gemm_bf16_kernel<1><<<gGemm512, 256, 0, stream>>>(b4, w_m1, m_b1, 1.0f,
      TOK, 512, 128, b0, nullptr, nullptr);
  gemm_bf16_kernel<3><<<gGemm128, 256, 0, stream>>>(b0, w_m2, m_b2, 1.0f,
      TOK, 128, 512, nullptr, x2, (float*)d_out);
}
